// ETLSTMTrain_35021163331744
// MI455X (gfx1250) — matrix operand order repaired and checked by reading, not compiled, not run on hardware
//
#include <hip/hip_runtime.h>

typedef __attribute__((ext_vector_type(16))) _Float16 v16h;
typedef __attribute__((ext_vector_type(8)))  _Float16 v8h;
typedef __attribute__((ext_vector_type(8)))  float    v8f;

#define TT 20
#define HH 64
#define DIN 40
#define KD 128

__device__ __forceinline__ v8f wmma_f16(v16h a, v16h b, v8f c) {
  // D = A(16x32) * B(32x16) + C, f32 accum
  return __builtin_amdgcn_wmma_f32_16x16x32_f16(false, a, false, b, (short)0, c,
                                                false, false);
}

// fast sigmoid/tanh: v_exp_f32 + v_rcp_f32 (no IEEE divide expansion)
__device__ __forceinline__ float fsigmoid(float x) {
  return __builtin_amdgcn_rcpf(1.0f + __expf(-x));
}
__device__ __forceinline__ float ftanh_(float x) {
  return fmaf(2.0f, fsigmoid(2.0f * x), -1.0f);
}

// A-fragment (16-bit, 16x32): lane L holds row m=L&15;
// elems i -> K = i + 16*(L>>4) (matches B-fragment order).
__device__ __forceinline__ v16h load_A_frag(const _Float16* row, int kb, int half) {
  const _Float16* p = row + kb * 32 + half * 16;
  v8h lo = *(const v8h*)(p);
  v8h hi = *(const v8h*)(p + 8);
  v16h a;
#pragma unroll
  for (int i = 0; i < 8; ++i) { a[i] = lo[i]; a[i + 8] = hi[i]; }
  return a;
}

// B-fragment (16-bit, 32x16): lane L holds col n=L&15; elems i -> K = i + 16*(L>>4).
// With W stored row-major (out x K) this is 16 contiguous halves (one 32B load).
__device__ __forceinline__ v16h load_B_frag(const _Float16* W, int row, int stride,
                                            int kb, int half) {
  return *(const v16h*)(W + (size_t)row * stride + kb * 32 + half * 16);
}

__device__ __forceinline__ void atomic_add_f32(float* p, float v) {
  __hip_atomic_fetch_add(p, v, __ATOMIC_RELAXED, __HIP_MEMORY_SCOPE_AGENT);
}

// ---------------------------------------------------------------------------
// Pack weights to f16: Wc = [W_ih | 0pad | W_hh] (256x128), Weo/Wn (64x128),
// WfcP = Wfc padded to 16x64, biasc = b_ih + b_hh.
// ---------------------------------------------------------------------------
__global__ __launch_bounds__(256) void prep_kernel(
    const float* __restrict__ W_ih, const float* __restrict__ W_hh,
    const float* __restrict__ b_ih, const float* __restrict__ b_hh,
    const float* __restrict__ Weo,  const float* __restrict__ Wn,
    const float* __restrict__ Wfc,
    _Float16* __restrict__ Wc, float* __restrict__ biasc,
    _Float16* __restrict__ Weo16, _Float16* __restrict__ Wn16,
    _Float16* __restrict__ WfcP) {
  int i = blockIdx.x * blockDim.x + threadIdx.x;
  if (i < 256 * KD) {
    int g = i >> 7, k = i & 127;
    float v = (k < DIN) ? W_ih[g * DIN + k]
                        : ((k < HH) ? 0.0f : W_hh[g * HH + (k - HH)]);
    Wc[i] = (_Float16)v;
  }
  if (i < 256) biasc[i] = b_ih[i] + b_hh[i];
  if (i < HH * KD) { Weo16[i] = (_Float16)Weo[i]; Wn16[i] = (_Float16)Wn[i]; }
  if (i < 16 * HH) {
    int nn = i >> 6, k = i & 63;
    WfcP[i] = (_Float16)((nn < 10) ? Wfc[nn * HH + k] : 0.0f);
  }
}

// ---------------------------------------------------------------------------
// Edge kernel: 16 edges per 128-thread workgroup (4 waves).
// LSTM over T steps (WMMA gate GEMM), then m = relu([h_src|e]@Weo^T+beo),
// scattered into agg with fp32 atomics.
// ---------------------------------------------------------------------------
__global__ __launch_bounds__(128) void edge_lstm_kernel(
    const float* __restrict__ features,
    const float* __restrict__ edge_features,
    const float* __restrict__ seq_times,
    const float* __restrict__ tw, const float* __restrict__ tb,
    const float* __restrict__ tw0, const float* __restrict__ tb0,
    const _Float16* __restrict__ Wc, const float* __restrict__ biasc,
    const _Float16* __restrict__ Weo16, const float* __restrict__ beo,
    const int* __restrict__ edge_len, const int* __restrict__ src_ids,
    const int* __restrict__ dst_ids,
    float* __restrict__ agg, int Etot) {
  __shared__ _Float16 xh[16][KD];   // [edge][ x(0..63, 40..63 zero) | h(64..127) ]
  __shared__ float s_tw[31], s_tb[31], s_t0[2];
  __shared__ int s_len[16], s_src[16], s_dst[16];
  __shared__ int s_tmax;

  const int tid  = threadIdx.x;
  const int wave = tid >> 5;
  const int lane = tid & 31;
  const int n    = lane & 15;
  const int half = lane >> 4;
  const int e0   = blockIdx.x * 16;
  const int hidx = wave * 16 + n;   // hidden column owned by this lane

  if (tid < 31) { s_tw[tid] = tw[tid]; s_tb[tid] = tb[tid]; }
  if (tid == 32) { s_t0[0] = tw0[0]; s_t0[1] = tb0[0]; }
  if (tid >= 64 && tid < 80) {
    int m = tid - 64, e = e0 + m;
    bool v = e < Etot;
    s_len[m] = v ? edge_len[e] : 0;
    s_src[m] = v ? src_ids[e]  : 0;
    s_dst[m] = v ? dst_ids[e]  : 0;
  }
  for (int i = tid; i < 16 * KD; i += 128) xh[i >> 7][i & 127] = (_Float16)0.0f;

  // Weight fragments: tile j covers gates j*64 + hidx (j: i,f,g,o).
  v16h Bg[4][4];
  float bj[4];
#pragma unroll
  for (int j = 0; j < 4; ++j) {
    int g = j * 64 + hidx;
    bj[j] = biasc[g];
#pragma unroll
    for (int kb = 0; kb < 4; ++kb) Bg[j][kb] = load_B_frag(Wc, g, KD, kb, half);
  }

  float cs[8], hs[8];
#pragma unroll
  for (int r = 0; r < 8; ++r) { cs[r] = 0.0f; hs[r] = 0.0f; }

  __syncthreads();
  if (tid == 0) {
    int mx = 0;
#pragma unroll
    for (int i = 0; i < 16; ++i) mx = max(mx, s_len[i]);
    s_tmax = mx;
  }
  __syncthreads();
  const int tmax = s_tmax;

  for (int t = 0; t < tmax; ++t) {
    // stage x_t = [edge_feat(8) | sin(tau*tw+tb)(31) | tau*tw0+tb0(1)]
    for (int i = tid; i < 16 * DIN; i += 128) {
      int m = i / DIN, k = i - m * DIN;
      int e = e0 + m;
      float val = 0.0f;
      if (e < Etot) {
        if (k < 8) {
          val = edge_features[((size_t)e * TT + t) * 8 + k];
        } else {
          float tau = seq_times[(size_t)e * TT + t];
          val = (k < 39) ? __sinf(fmaf(tau, s_tw[k - 8], s_tb[k - 8]))
                         : fmaf(tau, s_t0[0], s_t0[1]);
        }
      }
      xh[m][k] = (_Float16)val;
    }
    __syncthreads();

    v16h A[4];
#pragma unroll
    for (int kb = 0; kb < 4; ++kb) A[kb] = load_A_frag(&xh[n][0], kb, half);

    v8f acc[4];
#pragma unroll
    for (int j = 0; j < 4; ++j) {
      v8f c;
#pragma unroll
      for (int r = 0; r < 8; ++r) c[r] = bj[j];
#pragma unroll
      for (int kb = 0; kb < 4; ++kb) c = wmma_f16(A[kb], Bg[j][kb], c);
      acc[j] = c;
    }
    __syncthreads();  // A (h-region) fully consumed before h update below

    // branchless LSTM cell update; mask via select, unconditional LDS store
#pragma unroll
    for (int r = 0; r < 8; ++r) {
      int me = r + 8 * half;          // edge row of this accum element
      bool upd = t < s_len[me];
      float ig = fsigmoid(acc[0][r]);
      float fg = fsigmoid(acc[1][r]);
      float gg = ftanh_(acc[2][r]);
      float og = fsigmoid(acc[3][r]);
      float cn = fmaf(fg, cs[r], ig * gg);
      float hn = og * ftanh_(cn);
      cs[r] = upd ? cn : cs[r];
      hs[r] = upd ? hn : hs[r];
      xh[me][HH + hidx] = (_Float16)hs[r];
    }
  }
  __syncthreads();

  // stage h_src into cols 0..63 (e stays in cols 64..127)
  for (int i = tid; i < 16 * HH; i += 128) {
    int m = i >> 6, k = i & 63;
    int e = e0 + m;
    float val = (e < Etot) ? features[(size_t)s_src[m] * HH + k] : 0.0f;
    xh[m][k] = (_Float16)val;
  }
  __syncthreads();

  // m = relu([h_src | e] @ Weo^T + beo); scatter-add into agg[dst]
  {
    v16h A[4], Bm[4];
#pragma unroll
    for (int kb = 0; kb < 4; ++kb) A[kb]  = load_A_frag(&xh[n][0], kb, half);
#pragma unroll
    for (int kb = 0; kb < 4; ++kb) Bm[kb] = load_B_frag(Weo16, hidx, KD, kb, half);
    float bm = beo[hidx];
    v8f c;
#pragma unroll
    for (int r = 0; r < 8; ++r) c[r] = bm;
#pragma unroll
    for (int kb = 0; kb < 4; ++kb) c = wmma_f16(A[kb], Bm[kb], c);
#pragma unroll
    for (int r = 0; r < 8; ++r) {
      int me = r + 8 * half;
      if (e0 + me < Etot)
        atomic_add_f32(&agg[(size_t)s_dst[me] * HH + hidx], fmaxf(c[r], 0.0f));
    }
  }
}

// ---------------------------------------------------------------------------
// Dst kernel: 16 nodes / workgroup.
// sh_tmp = self_h @ WeoL^T + beo ; h = (agg - sh_tmp)*subg_norm ;
// h2 = relu([self_h|h] @ Wn^T + bn) ; out = h2 @ Wfc^T + bfc.
// ---------------------------------------------------------------------------
__global__ __launch_bounds__(128) void dst_kernel(
    const float* __restrict__ features,
    const float* __restrict__ subg_norm,
    const float* __restrict__ agg,
    const _Float16* __restrict__ Weo16, const float* __restrict__ beo,
    const _Float16* __restrict__ Wn16,  const float* __restrict__ bn,
    const _Float16* __restrict__ WfcP,  const float* __restrict__ bfc,
    float* __restrict__ out, int Ndst) {
  __shared__ _Float16 sh[16][KD];
  const int tid  = threadIdx.x;
  const int wave = tid >> 5;
  const int lane = tid & 31;
  const int n    = lane & 15;
  const int half = lane >> 4;
  const int n0   = blockIdx.x * 16;
  const int col  = wave * 16 + n;

  for (int i = tid; i < 16 * HH; i += 128) {
    int m = i >> 6, k = i & 63;
    int node = n0 + m;
    sh[m][k] = (_Float16)((node < Ndst) ? features[(size_t)node * HH + k] : 0.0f);
  }
  __syncthreads();

  // sh_tmp = self_h @ Weo[:, :64]^T + beo   (K = 64)
  v8f st;
  {
    v16h A0 = load_A_frag(&sh[n][0], 0, half);
    v16h A1 = load_A_frag(&sh[n][0], 1, half);
    v16h B0 = load_B_frag(Weo16, col, KD, 0, half);
    v16h B1 = load_B_frag(Weo16, col, KD, 1, half);
    float bb = beo[col];
#pragma unroll
    for (int r = 0; r < 8; ++r) st[r] = bb;
    st = wmma_f16(A0, B0, st);
    st = wmma_f16(A1, B1, st);
  }
#pragma unroll
  for (int r = 0; r < 8; ++r) {
    int m = r + 8 * half;
    int node = n0 + m;
    float hv = 0.0f;
    if (node < Ndst)
      hv = (agg[(size_t)node * HH + col] - st[r]) * subg_norm[node];
    sh[m][HH + col] = (_Float16)hv;
  }
  __syncthreads();

  // h2 = relu([self_h | h] @ Wn^T + bn)
  v8f h2;
  {
    v16h A[4], B[4];
#pragma unroll
    for (int kb = 0; kb < 4; ++kb) A[kb] = load_A_frag(&sh[n][0], kb, half);
#pragma unroll
    for (int kb = 0; kb < 4; ++kb) B[kb] = load_B_frag(Wn16, col, KD, kb, half);
    float bb = bn[col];
#pragma unroll
    for (int r = 0; r < 8; ++r) h2[r] = bb;
#pragma unroll
    for (int kb = 0; kb < 4; ++kb) h2 = wmma_f16(A[kb], B[kb], h2);
  }
  __syncthreads();
#pragma unroll
  for (int r = 0; r < 8; ++r) sh[r + 8 * half][col] = (_Float16)fmaxf(h2[r], 0.0f);
  __syncthreads();

  // out = h2 @ WfcP^T + bfc   (one 16x16 tile; 10 valid cols; wave 0 only)
  if (wave == 0) {
    v16h A0 = load_A_frag(&sh[n][0], 0, half);
    v16h A1 = load_A_frag(&sh[n][0], 1, half);
    v16h B0 = load_B_frag(WfcP, n, HH, 0, half);
    v16h B1 = load_B_frag(WfcP, n, HH, 1, half);
    float bb = (n < 10) ? bfc[n] : 0.0f;
    v8f c;
#pragma unroll
    for (int r = 0; r < 8; ++r) c[r] = bb;
    c = wmma_f16(A0, B0, c);
    c = wmma_f16(A1, B1, c);
    if (n < 10) {
#pragma unroll
      for (int r = 0; r < 8; ++r) {
        int node = n0 + r + 8 * half;
        if (node < Ndst) out[(size_t)node * 10 + n] = c[r];
      }
    }
  }
}

extern "C" void kernel_launch(void* const* d_in, const int* in_sizes, int n_in,
                              void* d_out, int out_size, void* d_ws, size_t ws_size,
                              hipStream_t stream) {
  const float* features      = (const float*)d_in[0];
  const float* edge_features = (const float*)d_in[1];
  const float* seq_times     = (const float*)d_in[2];
  const float* subg_norm     = (const float*)d_in[3];
  const float* tw   = (const float*)d_in[4];
  const float* tb   = (const float*)d_in[5];
  const float* tw0  = (const float*)d_in[6];
  const float* tb0  = (const float*)d_in[7];
  const float* W_ih = (const float*)d_in[8];
  const float* W_hh = (const float*)d_in[9];
  const float* b_ih = (const float*)d_in[10];
  const float* b_hh = (const float*)d_in[11];
  const float* Weo  = (const float*)d_in[12];
  const float* beo  = (const float*)d_in[13];
  const float* Wn   = (const float*)d_in[14];
  const float* bn   = (const float*)d_in[15];
  const float* Wfc  = (const float*)d_in[16];
  const float* bfc  = (const float*)d_in[17];
  const int* edge_len = (const int*)d_in[18];
  const int* src_ids  = (const int*)d_in[19];
  const int* dst_ids  = (const int*)d_in[20];

  const int E    = in_sizes[18];
  const int Ndst = in_sizes[3];

  char* ws = (char*)d_ws;
  size_t off = 0;
  _Float16* Wc    = (_Float16*)(ws + off); off += (size_t)256 * KD * 2;
  _Float16* Weo16 = (_Float16*)(ws + off); off += (size_t)HH * KD * 2;
  _Float16* Wn16  = (_Float16*)(ws + off); off += (size_t)HH * KD * 2;
  _Float16* WfcP  = (_Float16*)(ws + off); off += (size_t)16 * HH * 2;
  float* biasc    = (float*)(ws + off);    off += (size_t)256 * 4;
  float* agg      = (float*)(ws + off);    off += (size_t)Ndst * HH * 4;

  hipMemsetAsync(agg, 0, (size_t)Ndst * HH * 4, stream);
  prep_kernel<<<128, 256, 0, stream>>>(W_ih, W_hh, b_ih, b_hh, Weo, Wn, Wfc,
                                       Wc, biasc, Weo16, Wn16, WfcP);
  edge_lstm_kernel<<<(E + 15) / 16, 128, 0, stream>>>(
      features, edge_features, seq_times, tw, tb, tw0, tb0,
      Wc, biasc, Weo16, beo, edge_len, src_ids, dst_ids, agg, E);
  dst_kernel<<<(Ndst + 15) / 16, 128, 0, stream>>>(
      features, subg_norm, agg, Weo16, beo, Wn16, bn, WfcP, bfc,
      (float*)d_out, Ndst);
}
